// MaskDINOEncoder_19258633355701
// MI455X (gfx1250) — matrix operand order repaired and checked by reading, not compiled, not run on hardware
//
#include <hip/hip_runtime.h>
#include <hip/hip_bf16.h>
#include <stdint.h>

// ---------------- problem constants ----------------
constexpr int Bc   = 2;
constexpr int Cc   = 256;
constexpr int NHc  = 8;
constexpr int DHc  = 32;
constexpr int NLc  = 4;
constexpr int NPc  = 4;
constexpr int LAYERS = 6;
constexpr int DFFc = 1024;
constexpr int LQc  = 21760;          // 128^2 + 64^2 + 32^2 + 16^2
constexpr int BLc  = Bc * LQc;       // 43520 rows (multiple of 256)

// ---------------- WMMA types ----------------
typedef __attribute__((ext_vector_type(16))) __bf16 v16bf;
typedef __attribute__((ext_vector_type(8)))  float  v8f;

union FragAB { uint32_t u[8]; v16bf v; };

__device__ __forceinline__ uint16_t f2bf(float f) {
    uint32_t u = __float_as_uint(f);
    u += 0x7FFFu + ((u >> 16) & 1u);     // round-to-nearest-even
    return (uint16_t)(u >> 16);
}

// A fragment (16x32 bf16): lane lo = row M; K pairs per ISA layout.
__device__ __forceinline__ void loadFragA(FragAB& f, const uint16_t* __restrict__ arow,
                                          int k0, int hi) {
    #pragma unroll
    for (int j = 0; j < 8; ++j) {
        const int kb = (j < 4) ? (2*j + 8*hi) : (16 + 2*(j-4) + 8*hi);
        f.u[j] = *reinterpret_cast<const uint32_t*>(arow + k0 + kb);
    }
}

// B fragment (32x16 bf16) from BT [N,K]: lane lo = column N; K pairs per ISA layout (same as A).
__device__ __forceinline__ void loadFragB(FragAB& f, const uint16_t* __restrict__ bcol,
                                          int k0, int hi) {
    #pragma unroll
    for (int j = 0; j < 8; ++j) {
        const int kb = (j < 4) ? (2*j + 8*hi) : (16 + 2*(j-4) + 8*hi);
        f.u[j] = *reinterpret_cast<const uint32_t*>(bcol + k0 + kb);
    }
}

// ---------------- bf16 WMMA GEMM, 32x64 register tile per wave ----------------
// A: [M,K] bf16 row-major.  BT: [N,K] bf16 (B column-major).
// out = A*B + bias, optional relu, write f32 and/or bf16.
// M % 256 == 0, N % 64 == 0, K % 64 == 0 (true for all uses here).
// Two-stage software pipeline with statically-named stage registers (no dynamic
// array indexing, so fragments stay in VGPRs -- no scratch/LDS demotion).
__global__ __launch_bounds__(256)
void gemm_bf16_wmma(const uint16_t* __restrict__ A,
                    const uint16_t* __restrict__ BT,
                    const float*    __restrict__ bias,
                    float*          __restrict__ outF,
                    uint16_t*       __restrict__ outH,
                    int M, int N, int K, int relu)
{
    const int lane = threadIdx.x & 31;
    const int wave = threadIdx.x >> 5;
    const int mrow = blockIdx.x * 256 + wave * 32;   // 2 M-tiles per wave
    const int ncol = blockIdx.y * 64;                // 4 N-tiles per wave
    const int hi = lane >> 4;
    const int lo = lane & 15;

    v8f acc[2][4];
    #pragma unroll
    for (int mi = 0; mi < 2; ++mi)
        #pragma unroll
        for (int t = 0; t < 4; ++t)
            #pragma unroll
            for (int i = 0; i < 8; ++i) acc[mi][t][i] = 0.f;

    const uint16_t* arow0 = A + (size_t)(mrow + lo) * K;
    const uint16_t* arow1 = A + (size_t)(mrow + 16 + lo) * K;
    const uint16_t* bcol0 = BT + (size_t)(ncol +  0 + lo) * K;
    const uint16_t* bcol1 = BT + (size_t)(ncol + 16 + lo) * K;
    const uint16_t* bcol2 = BT + (size_t)(ncol + 32 + lo) * K;
    const uint16_t* bcol3 = BT + (size_t)(ncol + 48 + lo) * K;

    FragAB a0[2], b0[4], a1[2], b1[4];

    // stage 0 preload (k = 0)
    loadFragA(a0[0], arow0, 0, hi);
    loadFragA(a0[1], arow1, 0, hi);
    loadFragB(b0[0], bcol0, 0, hi);
    loadFragB(b0[1], bcol1, 0, hi);
    loadFragB(b0[2], bcol2, 0, hi);
    loadFragB(b0[3], bcol3, 0, hi);

    for (int k0 = 0; k0 < K; k0 += 64) {
        // prefetch stage 1 (k0+32, always in range since K % 64 == 0)
        const int k1 = k0 + 32;
        loadFragA(a1[0], arow0, k1, hi);
        loadFragA(a1[1], arow1, k1, hi);
        loadFragB(b1[0], bcol0, k1, hi);
        loadFragB(b1[1], bcol1, k1, hi);
        loadFragB(b1[2], bcol2, k1, hi);
        loadFragB(b1[3], bcol3, k1, hi);

        // compute stage 0
        #pragma unroll
        for (int t = 0; t < 4; ++t) {
            acc[0][t] = __builtin_amdgcn_wmma_f32_16x16x32_bf16(
                false, a0[0].v, false, b0[t].v, (short)0, acc[0][t], false, false);
            acc[1][t] = __builtin_amdgcn_wmma_f32_16x16x32_bf16(
                false, a0[1].v, false, b0[t].v, (short)0, acc[1][t], false, false);
        }

        // prefetch next stage 0 (k0+64) if any
        const int k2 = k0 + 64;
        if (k2 < K) {
            loadFragA(a0[0], arow0, k2, hi);
            loadFragA(a0[1], arow1, k2, hi);
            loadFragB(b0[0], bcol0, k2, hi);
            loadFragB(b0[1], bcol1, k2, hi);
            loadFragB(b0[2], bcol2, k2, hi);
            loadFragB(b0[3], bcol3, k2, hi);
        }

        // compute stage 1
        #pragma unroll
        for (int t = 0; t < 4; ++t) {
            acc[0][t] = __builtin_amdgcn_wmma_f32_16x16x32_bf16(
                false, a1[0].v, false, b1[t].v, (short)0, acc[0][t], false, false);
            acc[1][t] = __builtin_amdgcn_wmma_f32_16x16x32_bf16(
                false, a1[1].v, false, b1[t].v, (short)0, acc[1][t], false, false);
        }
    }

    #pragma unroll
    for (int t = 0; t < 4; ++t) {
        const int n = ncol + t*16 + lo;
        const float bv = bias ? bias[n] : 0.f;
        #pragma unroll
        for (int mi = 0; mi < 2; ++mi) {
            #pragma unroll
            for (int i = 0; i < 8; ++i) {
                const int m = mrow + mi*16 + i + 8*hi;
                float v = acc[mi][t][i] + bv;
                if (relu) v = v > 0.f ? v : 0.f;
                const size_t idx = (size_t)m * N + n;
                if (outF) outF[idx] = v;
                if (outH) outH[idx] = f2bf(v);
            }
        }
    }
}

// ---------------- weight convert: f32 [L,K,N] -> bf16 [L,N,K] ----------------
__global__ void wconv_kernel(const float* __restrict__ W, uint16_t* __restrict__ out,
                             int K, int N)
{
    size_t i = (size_t)blockIdx.x * blockDim.x + threadIdx.x;
    const size_t total = (size_t)LAYERS * K * N;
    if (i >= total) return;
    int l   = (int)(i / ((size_t)K * N));
    int rem = (int)(i - (size_t)l * K * N);
    int k = rem / N, n = rem - k * N;
    out[(size_t)l * K * N + (size_t)n * K + k] = f2bf(W[i]);
}

// ---------------- pack src levels -> x [B,LQ,C] f32 + bf16 ----------------
__global__ void pack_src_kernel(const float* __restrict__ s0, const float* __restrict__ s1,
                                const float* __restrict__ s2, const float* __restrict__ s3,
                                float* __restrict__ xf, uint16_t* __restrict__ xb)
{
    size_t i = (size_t)blockIdx.x * blockDim.x + threadIdx.x;
    if (i >= (size_t)BLc * Cc) return;
    int c  = (int)(i & 255);
    size_t bq = i >> 8;
    int b = (int)(bq / LQc);
    int q = (int)(bq - (size_t)b * LQc);
    const float* sp; int st, HW;
    if (q < 16384)      { sp = s0; st = 0;     HW = 16384; }
    else if (q < 20480) { sp = s1; st = 16384; HW = 4096;  }
    else if (q < 21504) { sp = s2; st = 20480; HW = 1024;  }
    else                { sp = s3; st = 21504; HW = 256;   }
    float v = sp[((size_t)b * Cc + c) * HW + (q - st)];
    xf[i] = v; xb[i] = f2bf(v);
}

// ---------------- pack pos levels + level_embed -> pos [B,LQ,C] f32 ----------------
__global__ void pack_pos_kernel(const float* __restrict__ p0, const float* __restrict__ p1,
                                const float* __restrict__ p2, const float* __restrict__ p3,
                                const float* __restrict__ le, float* __restrict__ posf)
{
    size_t i = (size_t)blockIdx.x * blockDim.x + threadIdx.x;
    if (i >= (size_t)BLc * Cc) return;
    int c  = (int)(i & 255);
    size_t bq = i >> 8;
    int b = (int)(bq / LQc);
    int q = (int)(bq - (size_t)b * LQc);
    const float* sp; int st, HW, lv;
    if (q < 16384)      { sp = p0; st = 0;     HW = 16384; lv = 0; }
    else if (q < 20480) { sp = p1; st = 16384; HW = 4096;  lv = 1; }
    else if (q < 21504) { sp = p2; st = 20480; HW = 1024;  lv = 2; }
    else                { sp = p3; st = 21504; HW = 256;   lv = 3; }
    posf[i] = sp[((size_t)b * Cc + c) * HW + (q - st)] + le[lv * Cc + c];
}

// ---------------- q = bf16(x + pos) ----------------
__global__ void addq_kernel(const float* __restrict__ xf, const float* __restrict__ posf,
                            uint16_t* __restrict__ qb)
{
    size_t i = (size_t)blockIdx.x * blockDim.x + threadIdx.x;
    if (i >= (size_t)BLc * Cc) return;
    qb[i] = f2bf(xf[i] + posf[i]);
}

// ---------------- softmax over 16 (per token, per head) ----------------
__global__ void softmax16_kernel(float* __restrict__ a)
{
    int t = blockIdx.x * blockDim.x + threadIdx.x;
    if (t >= BLc * NHc) return;
    int bq = t >> 3, h = t & 7;
    float* p = a + (size_t)bq * (NHc * NLc * NPc) + h * 16;
    float m = p[0];
    #pragma unroll
    for (int i = 1; i < 16; ++i) m = fmaxf(m, p[i]);
    float e[16], s = 0.f;
    #pragma unroll
    for (int i = 0; i < 16; ++i) { e[i] = __expf(p[i] - m); s += e[i]; }
    const float inv = 1.f / s;
    #pragma unroll
    for (int i = 0; i < 16; ++i) p[i] = e[i] * inv;
}

// ---------------- deformable attention gather ----------------
// one wave per (b,q,h); lane = channel d in [0,32). val: [B*LQ, C] f32.
__global__ __launch_bounds__(256)
void deform_kernel(const float* __restrict__ val, const float* __restrict__ off,
                   const float* __restrict__ attn, uint16_t* __restrict__ sampH)
{
    int gw = blockIdx.x * 8 + (threadIdx.x >> 5);
    if (gw >= BLc * NHc) return;
    const int lane = threadIdx.x & 31;
    const int bq = gw >> 3;
    const int h  = gw & 7;
    const int b  = bq / LQc;
    const int q  = bq - b * LQc;

    const int lh[4] = {128, 64, 32, 16};
    const int st[4] = {0, 16384, 20480, 21504};

    int lvq, r;
    if (q < 16384)      { lvq = 0; r = q;         }
    else if (q < 20480) { lvq = 1; r = q - 16384; }
    else if (q < 21504) { lvq = 2; r = q - 20480; }
    else                { lvq = 3; r = q - 21504; }
    const int wq = lh[lvq]; // square levels: w == h
    const float refx = ((r % wq) + 0.5f) / (float)wq;
    const float refy = ((r / wq) + 0.5f) / (float)lh[lvq];

    const float* offRow  = off  + (size_t)bq * 256 + h * (NLc * NPc * 2);
    const float* attnRow = attn + (size_t)bq * 128 + h * (NLc * NPc);

    float acc = 0.f;
    #pragma unroll
    for (int lv = 0; lv < 4; ++lv) {
        const int Hl = lh[lv], Wl = lh[lv], s0 = st[lv];
        const float* vbase = val + ((size_t)b * LQc + s0) * Cc + h * DHc + lane;
        const float invW = 1.f / (float)Wl, invH = 1.f / (float)Hl;
        for (int p = 0; p < 4; ++p) {
            const float ox = offRow[(lv * 4 + p) * 2 + 0];
            const float oy = offRow[(lv * 4 + p) * 2 + 1];
            const float aw = attnRow[lv * 4 + p];
            const float xx = (refx + ox * invW) * Wl - 0.5f;
            const float yy = (refy + oy * invH) * Hl - 0.5f;
            const float fx = floorf(xx), fy = floorf(yy);
            const float wx = xx - fx, wy = yy - fy;
            const int x0 = (int)fx, y0 = (int)fy;
            float s = 0.f;
            #pragma unroll
            for (int dy = 0; dy < 2; ++dy) {
                const int yi = y0 + dy;
                if (yi < 0 || yi >= Hl) continue;
                const float wyv = dy ? wy : (1.f - wy);
                #pragma unroll
                for (int dx = 0; dx < 2; ++dx) {
                    const int xi = x0 + dx;
                    if (xi < 0 || xi >= Wl) continue;
                    const float wxv = dx ? wx : (1.f - wx);
                    s += wxv * wyv * vbase[(size_t)(yi * Wl + xi) * Cc];
                }
            }
            acc += aw * s;
        }
    }
    sampH[(size_t)bq * Cc + h * DHc + lane] = f2bf(acc);
}

// ---------------- fused residual + LayerNorm, one wave per row (C=256) ----------------
// lane owns 8 contiguous channels; cross-lane reduce via shuffles (no LDS barriers).
__global__ __launch_bounds__(256)
void ln_residual_kernel(float* __restrict__ xf, const float* __restrict__ delta,
                        const float* __restrict__ g, const float* __restrict__ bb,
                        uint16_t* __restrict__ xb)
{
    const int lane = threadIdx.x & 31;
    const int row  = blockIdx.x * 8 + (threadIdx.x >> 5);
    if (row >= BLc) return;
    const size_t base = (size_t)row * Cc + lane * 8;

    float v[8];
    const float4 x0 = *reinterpret_cast<const float4*>(xf + base);
    const float4 x1 = *reinterpret_cast<const float4*>(xf + base + 4);
    const float4 d0 = *reinterpret_cast<const float4*>(delta + base);
    const float4 d1 = *reinterpret_cast<const float4*>(delta + base + 4);
    v[0] = x0.x + d0.x; v[1] = x0.y + d0.y; v[2] = x0.z + d0.z; v[3] = x0.w + d0.w;
    v[4] = x1.x + d1.x; v[5] = x1.y + d1.y; v[6] = x1.z + d1.z; v[7] = x1.w + d1.w;

    float s = 0.f;
    #pragma unroll
    for (int i = 0; i < 8; ++i) s += v[i];
    #pragma unroll
    for (int m = 16; m > 0; m >>= 1) s += __shfl_xor(s, m, 32);
    const float mu = s * (1.f / Cc);

    float vs = 0.f;
    #pragma unroll
    for (int i = 0; i < 8; ++i) { const float d = v[i] - mu; vs += d * d; }
    #pragma unroll
    for (int m = 16; m > 0; m >>= 1) vs += __shfl_xor(vs, m, 32);
    const float inv = rsqrtf(vs * (1.f / Cc) + 1e-5f);

    const int c = lane * 8;
    float y[8];
    #pragma unroll
    for (int i = 0; i < 8; ++i) y[i] = (v[i] - mu) * inv * g[c + i] + bb[c + i];

    float4 o0 = {y[0], y[1], y[2], y[3]};
    float4 o1 = {y[4], y[5], y[6], y[7]};
    *reinterpret_cast<float4*>(xf + base)     = o0;
    *reinterpret_cast<float4*>(xf + base + 4) = o1;

    uint4 hb;
    hb.x = (uint32_t)f2bf(y[0]) | ((uint32_t)f2bf(y[1]) << 16);
    hb.y = (uint32_t)f2bf(y[2]) | ((uint32_t)f2bf(y[3]) << 16);
    hb.z = (uint32_t)f2bf(y[4]) | ((uint32_t)f2bf(y[5]) << 16);
    hb.w = (uint32_t)f2bf(y[6]) | ((uint32_t)f2bf(y[7]) << 16);
    *reinterpret_cast<uint4*>(xb + base) = hb;
}

// ---------------- host side ----------------
extern "C" void kernel_launch(void* const* d_in, const int* in_sizes, int n_in,
                              void* d_out, int out_size, void* d_ws, size_t ws_size,
                              hipStream_t stream)
{
    (void)in_sizes; (void)n_in; (void)out_size; (void)ws_size;

    const float* src[4] = {(const float*)d_in[0], (const float*)d_in[1],
                           (const float*)d_in[2], (const float*)d_in[3]};
    const float* pos[4] = {(const float*)d_in[4], (const float*)d_in[5],
                           (const float*)d_in[6], (const float*)d_in[7]};
    const float* le    = (const float*)d_in[8];
    const float* Woff  = (const float*)d_in[9];
    const float* boff  = (const float*)d_in[10];
    const float* Wattn = (const float*)d_in[11];
    const float* battn = (const float*)d_in[12];
    const float* Wval  = (const float*)d_in[13];
    const float* bval  = (const float*)d_in[14];
    const float* Wout  = (const float*)d_in[15];
    const float* bout  = (const float*)d_in[16];
    const float* g1    = (const float*)d_in[17];
    const float* b1    = (const float*)d_in[18];
    const float* Wff1  = (const float*)d_in[19];
    const float* bff1  = (const float*)d_in[20];
    const float* Wff2  = (const float*)d_in[21];
    const float* bff2  = (const float*)d_in[22];
    const float* g2    = (const float*)d_in[23];
    const float* b2    = (const float*)d_in[24];

    const size_t BLC = (size_t)BLc * Cc;

    // ---- workspace carve ----
    char* w = (char*)d_ws;
    size_t o = 0;
    auto take = [&](size_t bytes) -> void* {
        void* p = (void*)(w + o);
        o += (bytes + 255) & ~(size_t)255;
        return p;
    };
    float*    xf      = (float*)take(BLC * 4);
    float*    posf    = (float*)take(BLC * 4);
    uint16_t* xb      = (uint16_t*)take(BLC * 2);
    uint16_t* qb      = (uint16_t*)take(BLC * 2);
    float*    offb    = (float*)take(BLC * 4);                 // [BL,256]
    float*    attnb   = (float*)take((size_t)BLc * 128 * 4);   // [BL,128]
    float*    valf    = (float*)take(BLC * 4);
    uint16_t* sampH   = (uint16_t*)take(BLC * 2);
    float*    attnout = (float*)take(BLC * 4);
    uint16_t* ffh     = (uint16_t*)take((size_t)BLc * DFFc * 2);
    float*    ffout   = (float*)take(BLC * 4);
    uint16_t* wbOff   = (uint16_t*)take((size_t)LAYERS * 256 * 256 * 2);
    uint16_t* wbAttn  = (uint16_t*)take((size_t)LAYERS * 256 * 128 * 2);
    uint16_t* wbVal   = (uint16_t*)take((size_t)LAYERS * 256 * 256 * 2);
    uint16_t* wbOut   = (uint16_t*)take((size_t)LAYERS * 256 * 256 * 2);
    uint16_t* wbFF1   = (uint16_t*)take((size_t)LAYERS * 256 * 1024 * 2);
    uint16_t* wbFF2   = (uint16_t*)take((size_t)LAYERS * 1024 * 256 * 2);

    // ---- convert + transpose weights to bf16 [L,N,K] ----
    auto wconv = [&](const float* W, uint16_t* dst, int K, int N) {
        size_t total = (size_t)LAYERS * K * N;
        wconv_kernel<<<(unsigned)((total + 255) / 256), 256, 0, stream>>>(W, dst, K, N);
    };
    wconv(Woff,  wbOff,  256, 256);
    wconv(Wattn, wbAttn, 256, 128);
    wconv(Wval,  wbVal,  256, 256);
    wconv(Wout,  wbOut,  256, 256);
    wconv(Wff1,  wbFF1,  256, 1024);
    wconv(Wff2,  wbFF2,  1024, 256);

    // ---- pack src / pos ----
    const unsigned nElemBlocks = (unsigned)((BLC + 255) / 256);
    pack_src_kernel<<<nElemBlocks, 256, 0, stream>>>(src[0], src[1], src[2], src[3], xf, xb);
    pack_pos_kernel<<<nElemBlocks, 256, 0, stream>>>(pos[0], pos[1], pos[2], pos[3], le, posf);

    auto gemm = [&](const uint16_t* A, const uint16_t* BT, const float* bias,
                    float* oF, uint16_t* oH, int M, int N, int K, int relu) {
        dim3 grid((unsigned)(M / 256), (unsigned)(N / 64));
        gemm_bf16_wmma<<<grid, 256, 0, stream>>>(A, BT, bias, oF, oH, M, N, K, relu);
    };

    // ---- encoder layers ----
    for (int l = 0; l < LAYERS; ++l) {
        const uint16_t* wOff = wbOff  + (size_t)l * 256 * 256;
        const uint16_t* wAtt = wbAttn + (size_t)l * 256 * 128;
        const uint16_t* wVal = wbVal  + (size_t)l * 256 * 256;
        const uint16_t* wOut = wbOut  + (size_t)l * 256 * 256;
        const uint16_t* wF1  = wbFF1  + (size_t)l * 256 * 1024;
        const uint16_t* wF2  = wbFF2  + (size_t)l * 1024 * 256;

        // q = bf16(x + pos)
        addq_kernel<<<nElemBlocks, 256, 0, stream>>>(xf, posf, qb);

        // sampling offsets / attention logits / values
        gemm(qb, wOff, boff  + l * 256, offb,  nullptr, BLc, 256, 256, 0);
        gemm(qb, wAtt, battn + l * 128, attnb, nullptr, BLc, 128, 256, 0);
        softmax16_kernel<<<(unsigned)((BLc * NHc + 255) / 256), 256, 0, stream>>>(attnb);
        gemm(xb, wVal, bval + l * 256, valf, nullptr, BLc, 256, 256, 0);

        // deformable gather -> bf16 sample output
        deform_kernel<<<(unsigned)(BLc * NHc / 8), 256, 0, stream>>>(valf, offb, attnb, sampH);

        // output projection + residual + LN1
        gemm(sampH, wOut, bout + l * 256, attnout, nullptr, BLc, 256, 256, 0);
        ln_residual_kernel<<<(unsigned)(BLc / 8), 256, 0, stream>>>(xf, attnout, g1 + l * 256, b1 + l * 256, xb);

        // FFN: relu(x@W1+b1)@W2+b2, residual + LN2
        gemm(xb, wF1, bff1 + l * 1024, nullptr, ffh, BLc, 1024, 256, 1);
        gemm(ffh, wF2, bff2 + l * 256, ffout, nullptr, BLc, 256, 1024, 0);
        ln_residual_kernel<<<(unsigned)(BLc / 8), 256, 0, stream>>>(xf, ffout, g2 + l * 256, b2 + l * 256, xb);
    }

    // ---- write result ----
    hipMemcpyAsync(d_out, xf, BLC * sizeof(float), hipMemcpyDeviceToDevice, stream);
}
